// FAVORCausalSelfAttention_21242908246344
// MI455X (gfx1250) — matrix operand order repaired and checked by reading, not compiled, not run on hardware
//
#include <hip/hip_runtime.h>
#include <hip/hip_bf16.h>
#include <math.h>

// ---------------------------------------------------------------------------
// FAVOR+ causal self-attention for MI455X (gfx1250), wave32.
// WMMA bf16 GEMMs with double-buffered GLOBAL_LOAD_ASYNC_TO_LDS staging,
// fp32 VALU phi (exp precision), chunked WMMA linear-attention scan.
// B=2, T=1024, C=1024, nh=16, d_h=64, m=128.
// ---------------------------------------------------------------------------

typedef __attribute__((ext_vector_type(16))) __bf16 v16bf;
typedef __attribute__((ext_vector_type(8)))  float  v8f;

constexpr int Bc = 2;
constexpr int Tc = 1024;
constexpr int Cc = 1024;
constexpr int NH = 16;
constexpr int DH = 64;    // head dim
constexpr int Mf = 128;   // FAVOR feature dim
constexpr int CT = 32;    // scan chunk length
constexpr float EPSc = 1e-6f;

// -------------------------- CDNA5 async-LDS helpers ------------------------
// GLOBAL_LOAD_ASYNC_TO_LDS_B128: per-lane 16B global -> LDS DMA, ASYNCcnt.
__device__ __forceinline__ void async_b128(const void* g, unsigned lds_off) {
  asm volatile("global_load_async_to_lds_b128 %0, %1, off"
               :: "v"(lds_off), "v"((unsigned long long)g)
               : "memory");
}
__device__ __forceinline__ void wait_async4() {
  asm volatile("s_wait_asynccnt 4" ::: "memory");
}
__device__ __forceinline__ void wait_async0() {
  asm volatile("s_wait_asynccnt 0" ::: "memory");
}
// Low 32 bits of a generic pointer into LDS aperture = LDS byte offset
// (ISA 10.2).
__device__ __forceinline__ unsigned lds_off32(const void* p) {
  return (unsigned)(unsigned long long)p;
}

// -------------------------- WMMA fragment helpers --------------------------
// Layouts per CDNA5 ISA 7.12.2 (wave32):
//  A 16x32 bf16: lane<16 -> row=lane,    K in {0..7}  U {16..23}
//                lane>=16 -> row=lane-16, K in {8..15} U {24..31}
//  B 32x16 bf16: lane -> col N=lane&15, K order same as A
//  C/D 16x16 f32: lane -> N=lane&15; vgpr e -> M = e + 8*(lane>>4)

__device__ __forceinline__ v16bf frag_a_bf16(const __bf16* p, int ld) {
  int lane = threadIdx.x & 31;
  int r = lane & 15;
  int kb = (lane < 16) ? 0 : 8;
  const __bf16* q = p + r * ld;
  v16bf a;
#pragma unroll
  for (int i = 0; i < 8; ++i) { a[i] = q[kb + i]; a[8 + i] = q[kb + 16 + i]; }
  return a;
}

__device__ __forceinline__ v16bf frag_a_f32(const float* p, int ld) {
  int lane = threadIdx.x & 31;
  int r = lane & 15;
  int kb = (lane < 16) ? 0 : 8;
  const float* q = p + r * ld;
  v16bf a;
#pragma unroll
  for (int i = 0; i < 8; ++i) {
    a[i]     = (__bf16)q[kb + i];
    a[8 + i] = (__bf16)q[kb + 16 + i];
  }
  return a;
}

__device__ __forceinline__ v16bf frag_b_bf16(const __bf16* p, int ld) {
  int lane = threadIdx.x & 31;
  int kb = (lane < 16) ? 0 : 8;
  const __bf16* q = p + (lane & 15);
  v16bf b;
#pragma unroll
  for (int i = 0; i < 8; ++i) { b[i] = q[(kb + i) * ld]; b[8 + i] = q[(kb + 16 + i) * ld]; }
  return b;
}

__device__ __forceinline__ v16bf frag_b_f32(const float* p, int ld) {
  int lane = threadIdx.x & 31;
  int kb = (lane < 16) ? 0 : 8;
  const float* q = p + (lane & 15);
  v16bf b;
#pragma unroll
  for (int i = 0; i < 8; ++i) { b[i] = (__bf16)q[(kb + i) * ld]; b[8 + i] = (__bf16)q[(kb + 16 + i) * ld]; }
  return b;
}

__device__ __forceinline__ v8f frag_c_load(const float* p, int ld) {
  int lane = threadIdx.x & 31;
  int n = lane & 15, half = lane >> 4;
  v8f c;
#pragma unroll
  for (int e = 0; e < 8; ++e) c[e] = p[(e + 8 * half) * ld + n];
  return c;
}

__device__ __forceinline__ void frag_c_store(float* p, int ld, v8f c) {
  int lane = threadIdx.x & 31;
  int n = lane & 15, half = lane >> 4;
#pragma unroll
  for (int e = 0; e < 8; ++e) p[(e + 8 * half) * ld + n] = c[e];
}

__device__ __forceinline__ v8f wmma_bf16(v16bf a, v16bf b, v8f c) {
  return __builtin_amdgcn_wmma_f32_16x16x32_bf16(false, a, false, b, (short)0,
                                                 c, false, false);
}

__device__ __forceinline__ v8f v8f_zero() {
  v8f z = {0.f, 0.f, 0.f, 0.f, 0.f, 0.f, 0.f, 0.f};
  return z;
}

// ---------------------------------------------------------------------------
// Elementwise f32 -> bf16 conversion (one pass; ~negligible vs 23.3 TB/s).
// ---------------------------------------------------------------------------
__global__ __launch_bounds__(256) void cvt_f32_bf16(const float* __restrict__ src,
                                                    __bf16* __restrict__ dst,
                                                    int n4) {
  int i = (blockIdx.x * 256 + threadIdx.x);
  if (i < n4) {
    float4 f = *(const float4*)(src + (size_t)i * 4);
    size_t o = (size_t)i * 4;
    dst[o + 0] = (__bf16)f.x;
    dst[o + 1] = (__bf16)f.y;
    dst[o + 2] = (__bf16)f.z;
    dst[o + 3] = (__bf16)f.w;
  }
}

// ---------------------------------------------------------------------------
// Async double-buffered GEMM: C[M,N] = A[M,K]*B[K,N] + bias[N].
// A,B bf16 in global; tiles DMA'd to LDS via global_load_async_to_lds_b128
// while WMMAs run on the other buffer. Block tile 64x64, BK=32, 4 waves.
// LDS strides are 16B multiples (async B128 alignment) with bank rotation.
// ---------------------------------------------------------------------------
__global__ __launch_bounds__(128) void gemm_bf16_async(const __bf16* __restrict__ A,
                                                       const __bf16* __restrict__ Bm,
                                                       const float* __restrict__ bias,
                                                       float* __restrict__ Cm,
                                                       int Mdim, int Ndim, int Kdim) {
  __shared__ __bf16 sA[2][64][40];  // 64x32 tile (+pad to 80B rows)
  __shared__ __bf16 sB[2][32][72];  // 32x64 tile (+pad to 144B rows)

  const int tid = threadIdx.x;
  const int w = tid >> 5;
  const int bm = blockIdx.x * 64;
  const int bn = blockIdx.y * 64;
  const int wm = (w >> 1) * 32;
  const int wn = (w & 1) * 32;

  v8f acc[2][2];
#pragma unroll
  for (int i = 0; i < 2; ++i)
#pragma unroll
    for (int j = 0; j < 2; ++j) acc[i][j] = v8f_zero();

  auto issue_tiles = [&](int buf, int k0) {
    // A tile: 64 rows x 4 chunks of 16B -> 256 chunks, 2 per thread
#pragma unroll
    for (int q = 0; q < 2; ++q) {
      int ci = tid + q * 128;
      int r = ci >> 2, c = ci & 3;
      async_b128(A + (size_t)(bm + r) * Kdim + k0 + c * 8,
                 lds_off32(&sA[buf][r][c * 8]));
    }
    // B tile: 32 rows x 8 chunks of 16B -> 256 chunks, 2 per thread
#pragma unroll
    for (int q = 0; q < 2; ++q) {
      int ci = tid + q * 128;
      int r = ci >> 3, c = ci & 7;
      async_b128(Bm + (size_t)(k0 + r) * Ndim + bn + c * 8,
                 lds_off32(&sB[buf][r][c * 8]));
    }
  };

  issue_tiles(0, 0);
  const int nIter = Kdim / 32;
  for (int it = 0; it < nIter; ++it) {
    const int cur = it & 1;
    if (it + 1 < nIter) {
      issue_tiles(cur ^ 1, (it + 1) * 32);
      wait_async4();   // all but the 4 newest done -> current buffer ready
    } else {
      wait_async0();
    }
    __syncthreads();

    v16bf a0 = frag_a_bf16(&sA[cur][wm][0], 40);
    v16bf a1 = frag_a_bf16(&sA[cur][wm + 16][0], 40);
    v16bf b0 = frag_b_bf16(&sB[cur][0][wn], 72);
    v16bf b1 = frag_b_bf16(&sB[cur][0][wn + 16], 72);
    acc[0][0] = wmma_bf16(a0, b0, acc[0][0]);
    acc[0][1] = wmma_bf16(a0, b1, acc[0][1]);
    acc[1][0] = wmma_bf16(a1, b0, acc[1][0]);
    acc[1][1] = wmma_bf16(a1, b1, acc[1][1]);
    __syncthreads();  // all waves done reading before next DMA overwrites
  }

  const int lane = tid & 31;
  const int n = lane & 15, half = lane >> 4;
#pragma unroll
  for (int i = 0; i < 2; ++i)
#pragma unroll
    for (int j = 0; j < 2; ++j) {
      int gm0 = bm + wm + i * 16;
      int gn = bn + wn + j * 16 + n;
      float bv = bias[gn];
#pragma unroll
      for (int e = 0; e < 8; ++e)
        Cm[(size_t)(gm0 + e + 8 * half) * Ndim + gn] = acc[i][j][e] + bv;
    }
}

// ---------------------------------------------------------------------------
// phi kernel: fp32 VALU (precision-sensitive exp). One block per (b,t).
// ---------------------------------------------------------------------------
__global__ __launch_bounds__(128) void phi_kernel(const float* __restrict__ qkv,
                                                  const float* __restrict__ omega,
                                                  float* __restrict__ phiQ,
                                                  float* __restrict__ phiK) {
  __shared__ float sO[DH * Mf];   // 32 KB, [d][m]
  __shared__ float sq[Cc];
  __shared__ float sk[Cc];
  __shared__ float nsh[2 * NH];

  const int tid = threadIdx.x;
  const int bt = blockIdx.x;
  const int b = bt >> 10;
  const int t = bt & 1023;
  const float scale = 0.125f;  // 1/sqrt(64)

  const float* row = qkv + (size_t)bt * (3 * Cc);
  for (int i = tid; i < DH * Mf; i += 128) sO[i] = omega[i];
  for (int i = tid; i < Cc; i += 128) {
    sq[i] = row[i] * scale;
    sk[i] = row[Cc + i] * scale;
  }
  __syncthreads();

  if (tid < 2 * NH) {
    const float* s = (tid < NH) ? sq : sk;
    int h = tid & (NH - 1);
    float acc = 0.f;
#pragma unroll 8
    for (int d = 0; d < DH; ++d) {
      float v = s[h * DH + d];
      acc += v * v;
    }
    nsh[tid] = 0.5f * acc;
  }
  __syncthreads();

  const float inv_sqrt_m = 0.08838834764831845f;  // 1/sqrt(128)
  for (int idx = tid; idx < 2 * NH * Mf; idx += 128) {
    int which = idx >> 11;
    int rem = idx & 2047;
    int h = rem >> 7;
    int m = rem & 127;
    const float* s = which ? sk : sq;
    float proj = 0.f;
#pragma unroll 8
    for (int d = 0; d < DH; ++d) proj += s[h * DH + d] * sO[d * Mf + m];
    float val = __expf(proj - nsh[which * NH + h]) * inv_sqrt_m;
    float* dst = which ? phiK : phiQ;
    dst[(((size_t)(b * NH + h) * Tc) + t) * Mf + m] = val;
  }
}

// ---------------------------------------------------------------------------
// Chunked causal linear-attention scan. One block (4 waves) per (b,h).
// State S (Mf x DH, f32) and Z (Mf) live in LDS; chunks of CT=32 timesteps.
//   P   = mask(phiq_c @ phik_c^T)              (32x32,  K=128)  WMMA
//   num = phiq_c @ S_prev + P @ V_c            (32x64)          WMMA
//   den = phiq_c . Z_prev + rowsum(P) + eps    (VALU)
//   S  += phik_c^T @ V_c ;  Z += colsum(phik)  (WMMA / VALU)
// Writes y directly as bf16 for the async output GEMM.
// ---------------------------------------------------------------------------
__global__ __launch_bounds__(128) void scan_kernel(const float* __restrict__ phiQ,
                                                   const float* __restrict__ phiK,
                                                   const float* __restrict__ qkv,
                                                   __bf16* __restrict__ Y) {
  __shared__ float  S[Mf][66];
  __shared__ float  Z[Mf];
  __shared__ float  P[CT][34];
  __shared__ float  den[CT];
  __shared__ __bf16 sPhiQ[CT][132];   // t x m
  __shared__ __bf16 sPhiKT[Mf][36];   // m x t
  __shared__ __bf16 sV[CT][68];       // t x d

  const int tid = threadIdx.x;
  const int w = tid >> 5;
  const int bh = blockIdx.x;
  const int b = bh / NH, h = bh % NH;

  const float* pq = phiQ + (size_t)bh * Tc * Mf;
  const float* pk = phiK + (size_t)bh * Tc * Mf;
  const float* vbase = qkv + 2 * Cc + h * DH;
  __bf16* ybase = Y + h * DH;

  for (int i = tid; i < Mf * 66; i += 128) (&S[0][0])[i] = 0.f;
  for (int i = tid; i < Mf; i += 128) Z[i] = 0.f;
  __syncthreads();

  for (int c = 0; c < Tc / CT; ++c) {
    const int t0 = c * CT;
    for (int i = tid; i < CT * Mf; i += 128) {
      int r = i >> 7, col = i & 127;
      sPhiQ[r][col] = (__bf16)pq[(size_t)(t0 + r) * Mf + col];
    }
    for (int i = tid; i < CT * Mf; i += 128) {
      int r = i >> 7, col = i & 127;
      sPhiKT[col][r] = (__bf16)pk[(size_t)(t0 + r) * Mf + col];
    }
    for (int i = tid; i < CT * DH; i += 128) {
      int r = i >> 6, col = i & 63;
      sV[r][col] = (__bf16)vbase[(size_t)(b * Tc + t0 + r) * (3 * Cc) + col];
    }
    __syncthreads();

    // P = phiq @ phikT (32x32, K=128): one fragment per wave
    {
      const int fm = (w >> 1) * 16, fn = (w & 1) * 16;
      v8f acc = v8f_zero();
#pragma unroll
      for (int kk = 0; kk < Mf; kk += 32) {
        v16bf a = frag_a_bf16(&sPhiQ[fm][kk], 132);
        v16bf bb = frag_b_bf16(&sPhiKT[kk][fn], 36);
        acc = wmma_bf16(a, bb, acc);
      }
      frag_c_store(&P[fm][fn], 34, acc);
    }
    __syncthreads();

    for (int i = tid; i < CT * CT; i += 128) {
      int r = i >> 5, s = i & 31;
      if (s > r) P[r][s] = 0.f;
    }
    __syncthreads();

    if (tid < CT) {
      float d0 = 0.f;
      const float* qr = pq + (size_t)(t0 + tid) * Mf;
#pragma unroll 8
      for (int m = 0; m < Mf; ++m) d0 += qr[m] * Z[m];
      float rs = 0.f;
      for (int s = 0; s <= tid; ++s) rs += P[tid][s];
      den[tid] = d0 + rs + EPSc;
    }
    __syncthreads();

    // num = phiq @ S_prev + P @ V ; y = num/den (bf16 out)
    {
      const int fr = (w >> 1) * 16;
#pragma unroll
      for (int fi = 0; fi < 2; ++fi) {
        const int fc = (w & 1) * 32 + fi * 16;
        v8f acc = v8f_zero();
#pragma unroll
        for (int kk = 0; kk < Mf; kk += 32) {
          v16bf a = frag_a_bf16(&sPhiQ[fr][kk], 132);
          v16bf bb = frag_b_f32(&S[kk][fc], 66);
          acc = wmma_bf16(a, bb, acc);
        }
        v16bf ap = frag_a_f32(&P[fr][0], 34);
        v16bf bv = frag_b_bf16(&sV[0][fc], 68);
        acc = wmma_bf16(ap, bv, acc);

        const int lane = tid & 31;
        const int n = lane & 15, half = lane >> 4;
#pragma unroll
        for (int e = 0; e < 8; ++e) {
          int tl = fr + e + 8 * half;
          ybase[(size_t)(b * Tc + t0 + tl) * Cc + fc + n] =
              (__bf16)(acc[e] / den[tl]);
        }
      }
    }
    __syncthreads();

    // S += phikT @ V (128x64, K=32)
#pragma unroll
    for (int rr = 0; rr < 2; ++rr) {
      const int fm = w * 32 + rr * 16;
      v16bf a = frag_a_bf16(&sPhiKT[fm][0], 36);
#pragma unroll
      for (int cc = 0; cc < 4; ++cc) {
        const int fd = cc * 16;
        v8f acc = frag_c_load(&S[fm][fd], 66);
        v16bf bb = frag_b_bf16(&sV[0][fd], 68);
        acc = wmma_bf16(a, bb, acc);
        frag_c_store(&S[fm][fd], 66, acc);
      }
    }
    if (tid < Mf) {
      float z = Z[tid];
      for (int r = 0; r < CT; ++r) z += pk[(size_t)(t0 + r) * Mf + tid];
      Z[tid] = z;
    }
    __syncthreads();
  }
}

// ---------------------------------------------------------------------------
extern "C" void kernel_launch(void* const* d_in, const int* in_sizes, int n_in,
                              void* d_out, int out_size, void* d_ws, size_t ws_size,
                              hipStream_t stream) {
  const float* x      = (const float*)d_in[0];  // (B,T,C)
  const float* W_attn = (const float*)d_in[1];  // (C,3C)
  const float* b_attn = (const float*)d_in[2];  // (3C)
  const float* W_proj = (const float*)d_in[3];  // (C,C)
  const float* b_proj = (const float*)d_in[4];  // (C)
  const float* omega  = (const float*)d_in[5];  // (DH,Mf)
  float* out = (float*)d_out;                   // (B,T,C)

  const size_t BT = (size_t)Bc * Tc;  // 2048
  // fp32 scratch
  float* qkv  = (float*)d_ws;                           // BT x 3C
  float* phiQ = qkv + BT * 3 * Cc;                      // B*NH*T*Mf
  float* phiK = phiQ + (size_t)Bc * NH * Tc * Mf;
  float* f32_end = phiK + (size_t)Bc * NH * Tc * Mf;
  // bf16 scratch (16B-aligned: all preceding sizes are 16B multiples)
  __bf16* xb  = (__bf16*)f32_end;                       // BT x C
  __bf16* Wab = xb + BT * Cc;                           // C x 3C
  __bf16* Wpb = Wab + (size_t)Cc * 3 * Cc;              // C x C
  __bf16* Yb  = Wpb + (size_t)Cc * Cc;                  // BT x C

  // 0) one-shot f32->bf16 conversions for the async GEMM operands
  {
    int n4x = (int)(BT * Cc / 4);
    cvt_f32_bf16<<<(n4x + 255) / 256, 256, 0, stream>>>(x, xb, n4x);
    int n4a = (int)((size_t)Cc * 3 * Cc / 4);
    cvt_f32_bf16<<<(n4a + 255) / 256, 256, 0, stream>>>(W_attn, Wab, n4a);
    int n4p = (int)((size_t)Cc * Cc / 4);
    cvt_f32_bf16<<<(n4p + 255) / 256, 256, 0, stream>>>(W_proj, Wpb, n4p);
  }
  // 1) qkv = x @ W_attn + b_attn   (async DMA + WMMA)
  dim3 g1(BT / 64, (3 * Cc) / 64);
  gemm_bf16_async<<<g1, 128, 0, stream>>>(xb, Wab, b_attn, qkv,
                                          (int)BT, 3 * Cc, Cc);
  // 2) phi features (fp32 VALU)
  phi_kernel<<<dim3(BT), 128, 0, stream>>>(qkv, omega, phiQ, phiK);
  // 3) chunked causal linear-attention scan (WMMA), emits bf16 y
  scan_kernel<<<dim3(Bc * NH), 128, 0, stream>>>(phiQ, phiK, qkv, Yb);
  // 4) out = y @ W_proj + b_proj   (async DMA + WMMA)
  dim3 g2(BT / 64, Cc / 64);
  gemm_bf16_async<<<g2, 128, 0, stream>>>(Yb, Wpb, b_proj, out,
                                          (int)BT, Cc, Cc);
}
